// ContrastiveLoss_69432441307668
// MI455X (gfx1250) — matrix operand order repaired and checked by reading, not compiled, not run on hardware
//
#include <hip/hip_runtime.h>
#include <hip/hip_bf16.h>
#include <stdint.h>

// ---------------------------------------------------------------------------
// ContrastiveLoss on MI455X (gfx1250, wave32).
//  Kernel 1: L2-normalize node embeddings, store as f16 in d_ws (12.8 MB,
//            sits in the 192 MB L2), zero the scalar output.
//  Kernel 2: one wave per 16-edge tile; 11 partner slots (dst + 10 negatives)
//            each evaluated with two chained v_wmma_f32_16x16x32_f16 (K=64);
//            the 16x16 tile diagonal (extracted register-side via shfl_xor,
//            no LDS) is the per-edge similarity. K==10 path fully unrolled:
//            22 WMMAs + straight-line logsumexp. Negative indices are hoisted
//            (5 Threefry calls, both output words used; umulhi range map).
// ---------------------------------------------------------------------------

#define DIMS 64
#define KMAX 10
#define TEMP_INV 2.0f   // 1 / TEMPERATURE (0.5)

typedef _Float16 v16h __attribute__((ext_vector_type(16)));
typedef _Float16 h8   __attribute__((ext_vector_type(8)));
typedef _Float16 h2   __attribute__((ext_vector_type(2)));
typedef float    v8f  __attribute__((ext_vector_type(8)));

__device__ __forceinline__ unsigned rotl32(unsigned x, unsigned n) {
    return (x << n) | (x >> (32u - n));
}

// Threefry-2x32-20, key = jax.random.key(42) ~ {0, 42}; both output words.
__device__ __forceinline__ uint2 threefry2x32(unsigned c0, unsigned c1) {
    const unsigned k0 = 0u, k1 = 42u;
    const unsigned k2 = k0 ^ k1 ^ 0x1BD11BDAu;
    const unsigned ks[3] = {k0, k1, k2};
    const unsigned R[8] = {13u, 15u, 26u, 6u, 17u, 29u, 16u, 24u};
    unsigned x0 = c0 + k0, x1 = c1 + k1;
#pragma unroll
    for (int s = 0; s < 5; ++s) {
#pragma unroll
        for (int r = 0; r < 4; ++r) {
            x0 += x1;
            x1 = rotl32(x1, R[(s & 1) * 4 + r]);
            x1 ^= x0;
        }
        x0 += ks[(s + 1) % 3];
        x1 += ks[(s + 2) % 3] + (unsigned)(s + 1);
    }
    return make_uint2(x0, x1);
}

// Map a uniform 32-bit word to [0, span) (one v_mul_hi_u32), then apply the
// reference's exclusion shifts for src/dst.
__device__ __forceinline__ int map_exclude(unsigned bits, unsigned span,
                                           int aa, int bb) {
    const int r  = (int)__umulhi(bits, span);
    const int r1 = r  + ((r  >= aa) ? 1 : 0);
    const int r2 = r1 + ((r1 >= bb) ? 1 : 0);
    return (aa == bb) ? r1 : r2;
}

// --------------------------- Kernel 1: normalize ---------------------------
// One wave per node row (D=64): lane l owns elements {2l, 2l+1}.
__global__ void __launch_bounds__(256)
contrastive_normalize_f16(const float* __restrict__ x,
                          _Float16* __restrict__ y,
                          int n_nodes,
                          float* __restrict__ out)
{
    if (blockIdx.x == 0 && threadIdx.x == 0) out[0] = 0.0f;

    const int wave = (int)((blockIdx.x * blockDim.x + threadIdx.x) >> 5);
    const int lane = (int)(threadIdx.x & 31u);
    if (wave >= n_nodes) return;

    const float2 v = *(const float2*)(x + (size_t)wave * DIMS + lane * 2);
    float ss = v.x * v.x + v.y * v.y;
#pragma unroll
    for (int off = 16; off > 0; off >>= 1)
        ss += __shfl_xor(ss, off, 32);

    const float invn = 1.0f / fmaxf(sqrtf(ss), 1e-12f);
    h2 o;
    o[0] = (_Float16)(v.x * invn);
    o[1] = (_Float16)(v.y * invn);
    *(h2*)(y + (size_t)wave * DIMS + lane * 2) = o;
}

// --------------------- partner similarity via WMMA -------------------------
// Scaled similarity between edge (e0+col)'s src and partner `node`.
// Lanes (m, m+16) both end up with edge m's value.
__device__ __forceinline__ float partner_sim(
    int node, const _Float16* __restrict__ embH, const v16h Afrag[2],
    int hi, int col)
{
    // B fragment (32x16 f16): lane holds column N=col; halves
    // K = kb*32 + hi*8 + {0..7, 16..23}  -> two 16B loads per lane per kb.
    const _Float16* brow = embH + (size_t)node * DIMS;
    v8f acc = {};
#pragma unroll
    for (int kb = 0; kb < 2; ++kb) {
        h8 b0 = *(const h8*)(brow + kb * 32 + hi * 8);
        h8 b1 = *(const h8*)(brow + kb * 32 + hi * 8 + 16);
        v16h Bfrag = __builtin_shufflevector(b0, b1, 0, 1, 2, 3, 4, 5, 6, 7,
                                             8, 9, 10, 11, 12, 13, 14, 15);
        acc = __builtin_amdgcn_wmma_f32_16x16x32_f16(
            false, Afrag[kb], false, Bfrag, (short)0, acc, false, false);
    }

    // Diagonal (M=col, N=col): held in VGPR col&7 at lane col (col<8) or
    // lane col+16 (col>=8). Local select + one shfl_xor(16), no LDS.
    const int idx = col & 7;
    float c = acc[7];
#pragma unroll
    for (int i = 0; i < 7; ++i)
        if (idx == i) c = acc[i];
    const float other  = __shfl_xor(c, 16, 32);
    const bool  holder = (hi == ((col >> 3) & 1));
    return (holder ? c : other) * TEMP_INV;
}

// --------------------------- Kernel 2: edge loss ---------------------------
__global__ void __launch_bounds__(256)
contrastive_edge_loss_wmma(const _Float16* __restrict__ embH,
                           const int* __restrict__ edge_index,
                           const int* __restrict__ k_neg_ptr,
                           float* __restrict__ out,
                           int n_nodes, int n_edges)
{
    const int lane = (int)(threadIdx.x & 31u);
    const int w    = (int)(threadIdx.x >> 5);
    const int tile = (int)blockIdx.x * 8 + w;
    const int e0   = tile * 16;
    if (e0 >= n_edges) return;          // wave-uniform exit (EXEC stays full)

    const int col = lane & 15;          // edge-within-tile / B column N
    const int hi  = lane >> 4;          // lane half selects K sub-range

    int e = e0 + col;
    const bool valid = (e < n_edges);
    if (!valid) e = n_edges - 1;

    int K = k_neg_ptr[0];
    K = K < 0 ? 0 : (K > KMAX ? KMAX : K);

    const int s_node = edge_index[e];
    const int d_node = edge_index[n_edges + e];
    const int aa = s_node < d_node ? s_node : d_node;
    const int bb = s_node < d_node ? d_node : s_node;
    const unsigned span = (unsigned)(n_nodes - 2);

    // ---- A fragments (ISA 7.12.2, 16-bit A 16x32):
    //  lanes 0-15  row M=lane,    halves = K{b..b+7, b+16..b+23}, b = kb*32
    //  lanes 16-31 row M=lane-16, same +8
    const _Float16* arow = embH + (size_t)s_node * DIMS;
    v16h Afrag[2];
#pragma unroll
    for (int kb = 0; kb < 2; ++kb) {
        const int base = kb * 32 + hi * 8;
        h8 c0 = *(const h8*)(arow + base);
        h8 c1 = *(const h8*)(arow + base + 16);
        Afrag[kb] = __builtin_shufflevector(c0, c1, 0, 1, 2, 3, 4, 5, 6, 7,
                                            8, 9, 10, 11, 12, 13, 14, 15);
    }

    float pos, lse;
    if (K == KMAX) {
        // Fast path (what the harness runs). Hoist all partner indices:
        // 5 Threefry calls, each feeding two negatives.
        int nodes[KMAX + 1];
        nodes[0] = d_node;
#pragma unroll
        for (int j = 0; j < KMAX / 2; ++j) {
            const unsigned ctr = (unsigned)e * (unsigned)(KMAX / 2)
                               + (unsigned)j;
            const uint2 bits = threefry2x32(0u, ctr);
            nodes[1 + 2 * j] = map_exclude(bits.x, span, aa, bb);
            nodes[2 + 2 * j] = map_exclude(bits.y, span, aa, bb);
        }

        // 22 WMMAs, sims in registers, two-pass logsumexp.
        float sims[KMAX + 1];
#pragma unroll
        for (int p = 0; p <= KMAX; ++p)
            sims[p] = partner_sim(nodes[p], embH, Afrag, hi, col);

        float m = sims[0];
#pragma unroll
        for (int p = 1; p <= KMAX; ++p) m = fmaxf(m, sims[p]);
        float ssum = 0.0f;
#pragma unroll
        for (int p = 0; p <= KMAX; ++p) ssum += __expf(sims[p] - m);
        pos = sims[0];
        lse = m + __logf(ssum);
    } else {
        // Generic fallback: dynamic trip count, online logsumexp.
        float m = -3.4e38f, ssum = 0.0f;
        pos = 0.0f;
        const int half = (K + 1) / 2;
        for (int p = 0; p <= K; ++p) {
            int node;
            if (p == 0) {
                node = d_node;
            } else {
                const int j = (p - 1) >> 1;
                const unsigned ctr = (unsigned)e * (unsigned)half + (unsigned)j;
                const uint2 bits = threefry2x32(0u, ctr);
                node = map_exclude(((p - 1) & 1) ? bits.y : bits.x,
                                   span, aa, bb);
            }
            const float s = partner_sim(node, embH, Afrag, hi, col);
            if (p == 0) pos = s;
            if (s > m) {
                ssum = ssum * __expf(m - s) + 1.0f;
                m = s;
            } else {
                ssum += __expf(s - m);
            }
        }
        lse = m + __logf(ssum);
    }

    float loss = 0.0f;
    if (hi == 0) {
        const float ratio = __expf(pos - lse);
        loss = -__logf(ratio + 1e-8f);
        if (!valid) loss = 0.0f;
    }
#pragma unroll
    for (int off = 16; off > 0; off >>= 1)
        loss += __shfl_xor(loss, off, 32);
    if (lane == 0) atomicAdd(out, loss);
}

// ------------------------------- launcher ---------------------------------
extern "C" void kernel_launch(void* const* d_in, const int* in_sizes, int n_in,
                              void* d_out, int out_size, void* d_ws, size_t ws_size,
                              hipStream_t stream)
{
    const float* node_emb   = (const float*)d_in[0];
    const int*   edge_index = (const int*)d_in[1];
    const int*   k_neg      = (const int*)d_in[2];
    float*       out        = (float*)d_out;

    const int n_nodes = in_sizes[0] / DIMS;
    const int n_edges = in_sizes[1] / 2;

    _Float16* embH = (_Float16*)d_ws;   // n_nodes * 64 f16 = 12.8 MB

    // Kernel 1: one wave per node.
    {
        const long threads = (long)n_nodes * 32;
        dim3 block(256);
        dim3 grid((unsigned)((threads + 255) / 256));
        contrastive_normalize_f16<<<grid, block, 0, stream>>>(
            node_emb, embH, n_nodes, out);
    }

    // Kernel 2: one wave per 16-edge tile, 8 waves per block.
    {
        const int tiles = (n_edges + 15) / 16;
        dim3 block(256);
        dim3 grid((unsigned)((tiles + 7) / 8));
        contrastive_edge_loss_wmma<<<grid, block, 0, stream>>>(
            embH, edge_index, k_neg, out, n_nodes, n_edges);
    }
}
